// TransformerTower_90838558310745
// MI455X (gfx1250) — matrix operand order repaired and checked by reading, not compiled, not run on hardware
//
#include <hip/hip_runtime.h>

namespace {

constexpr int L_ = 4, B_ = 2, S_ = 512, M_ = 512, D_ = 1024, H_ = 16, DK_ = 64, FF_ = 4096;
constexpr int T_ = S_ + M_;

typedef __bf16 bf16_t;
typedef bf16_t v16bf __attribute__((ext_vector_type(16)));
typedef bf16_t v8bf  __attribute__((ext_vector_type(8)));
typedef float  v8f   __attribute__((ext_vector_type(8)));

__device__ __forceinline__ unsigned short f2bu(float f) {
  union { float f; unsigned u; } v; v.f = f;
  unsigned u = v.u;
  u += 0x7fffu + ((u >> 16) & 1u);   // round-to-nearest-even bf16
  return (unsigned short)(u >> 16);
}

// A fragment: 16x32 bf16, lanes 0-15 rows M with K {0..7,16..23}, lanes 16-31 K {8..15,24..31}
__device__ __forceinline__ v16bf load_a_frag(const bf16_t* A, long lda, int mBase, int k0, int lane) {
  int m  = mBase + (lane & 15);
  int kb = (lane < 16) ? 0 : 8;
  const bf16_t* base = A + (long)m * lda + k0 + kb;
  v8bf lo = *(const v8bf*)(base);
  v8bf hi = *(const v8bf*)(base + 16);
  v16bf r;
#pragma unroll
  for (int i = 0; i < 8; ++i) { r[i] = lo[i]; r[i + 8] = hi[i]; }
  return r;
}

// B fragment from Bt (N x K row-major): lanes 0-15 cols N with K {0..7,16..23}, lanes 16-31 K {8..15,24..31}
__device__ __forceinline__ v16bf load_b_frag(const bf16_t* Bt, long ldb, int nBase, int k0, int lane) {
  int n  = nBase + (lane & 15);
  int kb = (lane < 16) ? 0 : 8;
  const bf16_t* base = Bt + (long)n * ldb + k0 + kb;
  v8bf lo = *(const v8bf*)(base);
  v8bf hi = *(const v8bf*)(base + 16);
  v16bf r;
#pragma unroll
  for (int i = 0; i < 8; ++i) { r[i] = lo[i]; r[i + 8] = hi[i]; }
  return r;
}

struct GemmP {
  const unsigned short* A;  long lda, sAb, sAh;   // row-major MxK (bf16 bits)
  const unsigned short* Bt; long ldb, sBb, sBh;   // row-major NxK (bf16 bits)
  float* C;                 long ldc, sCb, sCh;
  int Hsplit;               // batch z -> (z/Hsplit, z%Hsplit) for strides
  int M, N, K;
  float alpha;
  const float* bias;
  unsigned short* O1; unsigned short* O2;
  const float* B1;  const float* B2;
  int rowsPB;
  int mode;  // 0 plain(+bias,alpha) 1 mask 2 rel-scatter 3 bias+relu 4 q-dual 5 head-store 6 head-store-T
};

__global__ __launch_bounds__(256) void gemm_bf16_wmma(GemmP p) {
  const int z  = blockIdx.z;
  const int zb = z / p.Hsplit, zh = z % p.Hsplit;
  const bf16_t* A  = (const bf16_t*)p.A  + (long)zb * p.sAb + (long)zh * p.sAh;
  const bf16_t* Bt = (const bf16_t*)p.Bt + (long)zb * p.sBb + (long)zh * p.sBh;
  float* C = p.C + (long)zb * p.sCb + (long)zh * p.sCh;
  const int lane  = threadIdx.x & 31;
  const int w     = threadIdx.x >> 5;
  const int mBase = blockIdx.y * 128 + (w >> 1) * 32;
  const int nBase = blockIdx.x * 64  + (w & 1)  * 32;

  v8f acc[2][2];
#pragma unroll
  for (int i = 0; i < 2; ++i)
#pragma unroll
    for (int j = 0; j < 2; ++j)
#pragma unroll
      for (int e = 0; e < 8; ++e) acc[i][j][e] = 0.0f;

  for (int k0 = 0; k0 < p.K; k0 += 32) {
    v16bf a0 = load_a_frag(A, p.lda, mBase,      k0, lane);
    v16bf a1 = load_a_frag(A, p.lda, mBase + 16, k0, lane);
    v16bf b0 = load_b_frag(Bt, p.ldb, nBase,      k0, lane);
    v16bf b1 = load_b_frag(Bt, p.ldb, nBase + 16, k0, lane);
    if (k0 + 32 < p.K) {
      __builtin_prefetch(A  + (long)(mBase + (lane & 15)) * p.lda + k0 + 32, 0, 1);
      __builtin_prefetch(Bt + (long)(nBase + (lane & 15)) * p.ldb + k0 + 32, 0, 1);
    }
    acc[0][0] = __builtin_amdgcn_wmma_f32_16x16x32_bf16(false, a0, false, b0, (short)0, acc[0][0], false, false);
    acc[0][1] = __builtin_amdgcn_wmma_f32_16x16x32_bf16(false, a0, false, b1, (short)0, acc[0][1], false, false);
    acc[1][0] = __builtin_amdgcn_wmma_f32_16x16x32_bf16(false, a1, false, b0, (short)0, acc[1][0], false, false);
    acc[1][1] = __builtin_amdgcn_wmma_f32_16x16x32_bf16(false, a1, false, b1, (short)0, acc[1][1], false, false);
  }

  const int nLane = lane & 15, mOff = (lane >> 4) * 8;
#pragma unroll
  for (int i = 0; i < 2; ++i) {
#pragma unroll
    for (int j = 0; j < 2; ++j) {
      const int n = nBase + j * 16 + nLane;
#pragma unroll
      for (int r = 0; r < 8; ++r) {
        const int m = mBase + i * 16 + mOff + r;
        float v = acc[i][j][r];
        switch (p.mode) {
          case 0: {
            v = v * p.alpha + (p.bias ? p.bias[n] : 0.0f);
            C[(long)m * p.ldc + n] = v;
          } break;
          case 1: {  // content scores + mask (cols 0..M free, causal after)
            if (n > M_ + m) v += -1.0e6f;
            C[(long)m * p.ldc + n] = v;
          } break;
          case 2: {  // rel-shift scatter-add: logits[m][n - (S-1) + m] += v
            int t = n - (S_ - 1) + m;
            if ((unsigned)t < (unsigned)p.N) atomicAdd(&C[(long)m * p.ldc + t], v);
          } break;
          case 3: {  // FFN1: bias + relu
            v = fmaxf(v + p.bias[n], 0.0f);
            C[(long)m * p.ldc + n] = v;
          } break;
          case 4: {  // q: scale, add r_w_bias / r_r_bias, store [B,H,S,DK] bf16 twice
            v *= p.alpha;
            int b = m / p.rowsPB, s = m % p.rowsPB;
            int h = n >> 6, dk = n & 63;
            long o = (((long)(b * H_ + h)) * p.rowsPB + s) * DK_ + dk;
            p.O1[o] = f2bu(v + p.B1[h * DK_ + dk]);
            p.O2[o] = f2bu(v + p.B2[h * DK_ + dk]);
          } break;
          case 5: {  // k / r: store [B,H,T,DK] bf16
            int b = m / p.rowsPB, t = m % p.rowsPB;
            int h = n >> 6, dk = n & 63;
            p.O1[(((long)(b * H_ + h)) * p.rowsPB + t) * DK_ + dk] = f2bu(v);
          } break;
          case 6: {  // v: store transposed [B,H,DK,T] bf16
            int b = m / p.rowsPB, t = m % p.rowsPB;
            int h = n >> 6, dk = n & 63;
            p.O1[(((long)(b * H_ + h)) * DK_ + dk) * (long)p.rowsPB + t] = f2bu(v);
          } break;
        }
      }
    }
  }
}

__global__ __launch_bounds__(256) void softmax_rows(float* logits) {
  __shared__ float sm[8];
  const long row = blockIdx.x;
  float* rp = logits + row * (long)T_;
  const int tid = threadIdx.x;
  float v[4];
#pragma unroll
  for (int i = 0; i < 4; ++i) v[i] = rp[tid + i * 256];
  float mx = fmaxf(fmaxf(v[0], v[1]), fmaxf(v[2], v[3]));
#pragma unroll
  for (int m = 16; m > 0; m >>= 1) mx = fmaxf(mx, __shfl_xor(mx, m, 32));
  if ((tid & 31) == 0) sm[tid >> 5] = mx;
  __syncthreads();
  mx = sm[0];
#pragma unroll
  for (int i = 1; i < 8; ++i) mx = fmaxf(mx, sm[i]);
  __syncthreads();
  float s = 0.0f;
#pragma unroll
  for (int i = 0; i < 4; ++i) { v[i] = __expf(v[i] - mx); s += v[i]; }
#pragma unroll
  for (int m = 16; m > 0; m >>= 1) s += __shfl_xor(s, m, 32);
  if ((tid & 31) == 0) sm[tid >> 5] = s;
  __syncthreads();
  s = 0.0f;
#pragma unroll
  for (int i = 0; i < 8; ++i) s += sm[i];
  const float inv = 1.0f / s;
  __syncthreads();  // all fp32 reads done -> safe to overwrite row head with bf16
  unsigned short* wp = (unsigned short*)rp;
#pragma unroll
  for (int i = 0; i < 4; ++i) wp[tid + i * 256] = f2bu(v[i] * inv);
}

__global__ __launch_bounds__(256) void add_layernorm(float* h, const float* y,
                                                     const float* g, const float* b) {
  __shared__ float sm[8], sm2[8];
  const long row = blockIdx.x;
  float* hp = h + row * (long)D_;
  const float* yp = y + row * (long)D_;
  const int tid = threadIdx.x;
  float v[4]; float s = 0.0f, s2 = 0.0f;
#pragma unroll
  for (int i = 0; i < 4; ++i) {
    v[i] = hp[tid + i * 256] + yp[tid + i * 256];
    s += v[i]; s2 += v[i] * v[i];
  }
#pragma unroll
  for (int m = 16; m > 0; m >>= 1) { s += __shfl_xor(s, m, 32); s2 += __shfl_xor(s2, m, 32); }
  if ((tid & 31) == 0) { sm[tid >> 5] = s; sm2[tid >> 5] = s2; }
  __syncthreads();
  s = 0.0f; s2 = 0.0f;
#pragma unroll
  for (int i = 0; i < 8; ++i) { s += sm[i]; s2 += sm2[i]; }
  const float mu   = s / (float)D_;
  const float var  = s2 / (float)D_ - mu * mu;
  const float rstd = rsqrtf(var + 1e-5f);
#pragma unroll
  for (int i = 0; i < 4; ++i) {
    int c = tid + i * 256;
    hp[c] = (v[i] - mu) * rstd * g[c] + b[c];
  }
}

__global__ void tobf16_kernel(const float* in, unsigned short* out, long n) {
  long i = blockIdx.x * (long)blockDim.x + threadIdx.x;
  long stride = (long)gridDim.x * blockDim.x;
  for (; i < n; i += stride) out[i] = f2bu(in[i]);
}

__global__ void build_kv_kernel(const float* mem, const float* h, unsigned short* kvb) {
  long i = blockIdx.x * (long)blockDim.x + threadIdx.x;
  long stride = (long)gridDim.x * blockDim.x;
  const long n = (long)B_ * T_ * D_;
  for (; i < n; i += stride) {
    long b = i / ((long)T_ * D_);
    long rem = i - b * (long)T_ * D_;
    long t = rem / D_, d = rem - t * D_;
    float v = (t < M_) ? mem[(b * M_ + t) * D_ + d]
                       : h[(b * S_ + (t - M_)) * D_ + d];
    kvb[i] = f2bu(v);
  }
}

__global__ void pos_kernel(unsigned short* posb) {
  long i = blockIdx.x * (long)blockDim.x + threadIdx.x;
  long stride = (long)gridDim.x * blockDim.x;
  const long n = (long)T_ * D_;
  for (; i < n; i += stride) {
    int t = (int)(i / D_), d = (int)(i % D_);
    float pv = (float)(T_ - 1 - t);
    int dd = (d < D_ / 2) ? d : d - D_ / 2;
    float inv = __expf(-(2.0f * (float)dd / (float)D_) * 9.21034037198f);  // ln(10000)
    float ang = pv * inv;
    posb[i] = f2bu((d < D_ / 2) ? __sinf(ang) : __cosf(ang));
  }
}

// Wt[n][k] = bf16(W[k][n]); W is KxN fp32
__global__ __launch_bounds__(256) void wtrans_kernel(const float* W, unsigned short* Wt, int K, int N) {
  __shared__ float tile[32][33];
  const int k0 = blockIdx.y * 32, n0 = blockIdx.x * 32;
#pragma unroll
  for (int r = 0; r < 4; ++r)
    tile[threadIdx.y + r * 8][threadIdx.x] =
        W[(long)(k0 + threadIdx.y + r * 8) * N + n0 + threadIdx.x];
  __syncthreads();
#pragma unroll
  for (int r = 0; r < 4; ++r)
    Wt[(long)(n0 + threadIdx.y + r * 8) * K + k0 + threadIdx.x] =
        f2bu(tile[threadIdx.x][threadIdx.y + r * 8]);
}

}  // namespace

extern "C" void kernel_launch(void* const* d_in, const int* in_sizes, int n_in,
                              void* d_out, int out_size, void* d_ws, size_t ws_size,
                              hipStream_t stream) {
  (void)in_sizes; (void)n_in; (void)out_size; (void)ws_size;
  const float* x    = (const float*)d_in[0];
  const float* mem  = (const float*)d_in[1];
  const float* Wq   = (const float*)d_in[2];
  const float* Wk   = (const float*)d_in[3];
  const float* Wv   = (const float*)d_in[4];
  const float* Wr   = (const float*)d_in[5];
  const float* Wo   = (const float*)d_in[6];
  const float* rwb  = (const float*)d_in[7];
  const float* rrb  = (const float*)d_in[8];
  const float* ln1g = (const float*)d_in[9];
  const float* ln1b = (const float*)d_in[10];
  const float* ln2g = (const float*)d_in[11];
  const float* ln2b = (const float*)d_in[12];
  const float* W1   = (const float*)d_in[13];
  const float* b1   = (const float*)d_in[14];
  const float* W2   = (const float*)d_in[15];
  const float* b2   = (const float*)d_in[16];

  char* wsp = (char*)d_ws;
  auto carve = [&](size_t bytes) { char* p = wsp; wsp += (bytes + 255) & ~(size_t)255; return p; };
  unsigned short* posb = (unsigned short*)carve((size_t)T_ * D_ * 2);
  float*          h    = (float*)carve((size_t)B_ * S_ * D_ * 4);
  unsigned short* hb   = (unsigned short*)carve((size_t)B_ * S_ * D_ * 2);
  unsigned short* kvb  = (unsigned short*)carve((size_t)B_ * T_ * D_ * 2);
  unsigned short* wbuf = (unsigned short*)carve((size_t)D_ * FF_ * 2);
  unsigned short* qa   = (unsigned short*)carve((size_t)B_ * H_ * S_ * DK_ * 2);
  unsigned short* qb   = (unsigned short*)carve((size_t)B_ * H_ * S_ * DK_ * 2);
  unsigned short* kk   = (unsigned short*)carve((size_t)B_ * H_ * T_ * DK_ * 2);
  unsigned short* vt   = (unsigned short*)carve((size_t)B_ * H_ * T_ * DK_ * 2);
  unsigned short* rr   = (unsigned short*)carve((size_t)H_ * T_ * DK_ * 2);
  float*          logits = (float*)carve((size_t)B_ * H_ * S_ * T_ * 4);
  float*          y    = (float*)carve((size_t)B_ * S_ * D_ * 4);
  unsigned short* yb   = (unsigned short*)carve((size_t)B_ * S_ * D_ * 2);
  float*          f1   = (float*)carve((size_t)B_ * S_ * FF_ * 4);
  unsigned short* f1b  = (unsigned short*)carve((size_t)B_ * S_ * FF_ * 2);

  auto launch_gemm = [&](const GemmP& p, int Z) {
    dim3 g(p.N / 64, p.M / 128, Z);
    gemm_bf16_wmma<<<g, dim3(256), 0, stream>>>(p);
  };
  auto baseP = []() { GemmP p{}; p.Hsplit = 1; p.alpha = 1.0f; return p; };

  hipMemcpyAsync(h, x, (size_t)B_ * S_ * D_ * 4, hipMemcpyDeviceToDevice, stream);
  pos_kernel<<<1024, 256, 0, stream>>>(posb);

  for (int l = 0; l < L_; ++l) {
    const float* Wq_l = Wq + (size_t)l * D_ * D_;
    const float* Wk_l = Wk + (size_t)l * D_ * D_;
    const float* Wv_l = Wv + (size_t)l * D_ * D_;
    const float* Wr_l = Wr + (size_t)l * D_ * D_;
    const float* Wo_l = Wo + (size_t)l * D_ * D_;
    const float* W1_l = W1 + (size_t)l * D_ * FF_;
    const float* W2_l = W2 + (size_t)l * FF_ * D_;

    tobf16_kernel<<<1024, 256, 0, stream>>>(h, hb, (long)B_ * S_ * D_);
    build_kv_kernel<<<2048, 256, 0, stream>>>(mem + (size_t)l * B_ * M_ * D_, h, kvb);

    // ---- Q (scaled, dual-bias, head layout) ----
    wtrans_kernel<<<dim3(D_ / 32, D_ / 32), dim3(32, 8), 0, stream>>>(Wq_l, wbuf, D_, D_);
    { GemmP p = baseP();
      p.A = hb; p.lda = D_; p.Bt = wbuf; p.ldb = D_; p.C = logits; p.ldc = T_;
      p.M = B_ * S_; p.N = D_; p.K = D_; p.alpha = 0.125f;  // DK^-0.5
      p.O1 = qa; p.O2 = qb;
      p.B1 = rwb + (size_t)l * H_ * DK_; p.B2 = rrb + (size_t)l * H_ * DK_;
      p.rowsPB = S_; p.mode = 4; launch_gemm(p, 1); }

    // ---- K ----
    wtrans_kernel<<<dim3(D_ / 32, D_ / 32), dim3(32, 8), 0, stream>>>(Wk_l, wbuf, D_, D_);
    { GemmP p = baseP();
      p.A = kvb; p.lda = D_; p.Bt = wbuf; p.ldb = D_; p.C = logits; p.ldc = T_;
      p.M = B_ * T_; p.N = D_; p.K = D_;
      p.O1 = kk; p.rowsPB = T_; p.mode = 5; launch_gemm(p, 1); }

    // ---- V (stored transposed) ----
    wtrans_kernel<<<dim3(D_ / 32, D_ / 32), dim3(32, 8), 0, stream>>>(Wv_l, wbuf, D_, D_);
    { GemmP p = baseP();
      p.A = kvb; p.lda = D_; p.Bt = wbuf; p.ldb = D_; p.C = logits; p.ldc = T_;
      p.M = B_ * T_; p.N = D_; p.K = D_;
      p.O1 = vt; p.rowsPB = T_; p.mode = 6; launch_gemm(p, 1); }

    // ---- R = pos @ Wr ----
    wtrans_kernel<<<dim3(D_ / 32, D_ / 32), dim3(32, 8), 0, stream>>>(Wr_l, wbuf, D_, D_);
    { GemmP p = baseP();
      p.A = posb; p.lda = D_; p.Bt = wbuf; p.ldb = D_; p.C = logits; p.ldc = T_;
      p.M = T_; p.N = D_; p.K = D_;
      p.O1 = rr; p.rowsPB = T_; p.mode = 5; launch_gemm(p, 1); }

    // ---- content scores + mask ----
    { GemmP p = baseP();
      p.A = qa; p.lda = DK_; p.sAb = (long)S_ * DK_;
      p.Bt = kk; p.ldb = DK_; p.sBb = (long)T_ * DK_;
      p.C = logits; p.ldc = T_; p.sCb = (long)S_ * T_;
      p.M = S_; p.N = T_; p.K = DK_; p.mode = 1; launch_gemm(p, B_ * H_); }

    // ---- relative scores, scatter-added with rel-shift ----
    { GemmP p = baseP();
      p.Hsplit = H_;
      p.A = qb; p.lda = DK_; p.sAb = (long)H_ * S_ * DK_; p.sAh = (long)S_ * DK_;
      p.Bt = rr; p.ldb = DK_; p.sBb = 0; p.sBh = (long)T_ * DK_;
      p.C = logits; p.ldc = T_; p.sCb = (long)H_ * S_ * T_; p.sCh = (long)S_ * T_;
      p.M = S_; p.N = T_; p.K = DK_; p.mode = 2; launch_gemm(p, B_ * H_); }

    softmax_rows<<<B_ * H_ * S_, 256, 0, stream>>>(logits);

    // ---- attn = P @ V  (P read as bf16 in-place from logits rows) ----
    { GemmP p = baseP();
      p.Hsplit = H_;
      p.A = (const unsigned short*)logits; p.lda = 2 * T_;
      p.sAb = (long)H_ * S_ * 2 * T_; p.sAh = (long)S_ * 2 * T_;
      p.Bt = vt; p.ldb = T_; p.sBb = (long)H_ * DK_ * T_; p.sBh = (long)DK_ * T_;
      p.C = y; p.ldc = D_; p.sCb = (long)S_ * D_; p.sCh = DK_;
      p.M = S_; p.N = DK_; p.K = T_; p.mode = 0; launch_gemm(p, B_ * H_); }

    // ---- output projection ----
    tobf16_kernel<<<1024, 256, 0, stream>>>(y, yb, (long)B_ * S_ * D_);
    wtrans_kernel<<<dim3(D_ / 32, D_ / 32), dim3(32, 8), 0, stream>>>(Wo_l, wbuf, D_, D_);
    { GemmP p = baseP();
      p.A = yb; p.lda = D_; p.Bt = wbuf; p.ldb = D_;
      p.C = y; p.ldc = D_; p.M = B_ * S_; p.N = D_; p.K = D_; p.mode = 0; launch_gemm(p, 1); }

    add_layernorm<<<B_ * S_, 256, 0, stream>>>(h, y, ln1g + (size_t)l * D_, ln1b + (size_t)l * D_);

    // ---- FFN ----
    tobf16_kernel<<<1024, 256, 0, stream>>>(h, hb, (long)B_ * S_ * D_);
    wtrans_kernel<<<dim3(FF_ / 32, D_ / 32), dim3(32, 8), 0, stream>>>(W1_l, wbuf, D_, FF_);
    { GemmP p = baseP();
      p.A = hb; p.lda = D_; p.Bt = wbuf; p.ldb = D_;
      p.C = f1; p.ldc = FF_; p.M = B_ * S_; p.N = FF_; p.K = D_;
      p.bias = b1 + (size_t)l * FF_; p.mode = 3; launch_gemm(p, 1); }

    tobf16_kernel<<<4096, 256, 0, stream>>>(f1, f1b, (long)B_ * S_ * FF_);
    wtrans_kernel<<<dim3(D_ / 32, FF_ / 32), dim3(32, 8), 0, stream>>>(W2_l, wbuf, FF_, D_);
    { GemmP p = baseP();
      p.A = f1b; p.lda = FF_; p.Bt = wbuf; p.ldb = FF_;
      p.C = y; p.ldc = D_; p.M = B_ * S_; p.N = D_; p.K = FF_;
      p.bias = b2 + (size_t)l * D_; p.mode = 0; launch_gemm(p, 1); }

    add_layernorm<<<B_ * S_, 256, 0, stream>>>(h, y, ln2g + (size_t)l * D_, ln2b + (size_t)l * D_);
  }

  hipMemcpyAsync(d_out, h, (size_t)B_ * S_ * D_ * 4, hipMemcpyDeviceToDevice, stream);
}
